// DecoderBlock_7980049236531
// MI455X (gfx1250) — matrix operand order repaired and checked by reading, not compiled, not run on hardware
//
#include <hip/hip_runtime.h>
#include <hip/hip_bf16.h>
#include <math.h>

// ---------------------------------------------------------------------------
// MI455X (gfx1250) decoder block: bf16 WMMA GEMMs + flash attention + TDM
// ---------------------------------------------------------------------------

typedef __attribute__((ext_vector_type(16))) __bf16 bf16x16;
typedef __attribute__((ext_vector_type(8)))  float  floatx8;
typedef __attribute__((ext_vector_type(4)))  unsigned int uint32x4;
typedef __attribute__((ext_vector_type(8)))  int    int32x8;
typedef __attribute__((ext_vector_type(4)))  int    int32x4;

#if defined(__has_builtin)
#if __has_builtin(__builtin_amdgcn_tensor_load_to_lds) && \
    __has_builtin(__builtin_amdgcn_s_wait_tensorcnt)
#define HAVE_TDM 1
#endif
#endif
#ifndef HAVE_TDM
#define HAVE_TDM 0
#endif

#define D_MODEL 1024
#define NHEAD   16
#define HDIM    64
#define SEQ     1024
#define BATCH   8
#define MROWS   (BATCH * SEQ)   /* 8192 */
#define FFDIM   4096

__device__ __forceinline__ floatx8 zero8() {
  floatx8 v;
#pragma unroll
  for (int i = 0; i < 8; ++i) v[i] = 0.0f;
  return v;
}

__device__ __forceinline__ __bf16 bf_from_bits(unsigned short u) {
  union { unsigned short s; __bf16 b; } cv;
  cv.s = u;
  return cv.b;
}

#if HAVE_TDM
// ---------------------------------------------------------------------------
// TDM: issue TENSOR_LOAD_TO_LDS of a 2D bf16 tile (tile_d1 rows of tile_d0
// elems, row stride td0_stride elems) into LDS at byte offset `lds`, with
// hardware padding of 4 DWORDs after every 16 DWORDs -> 40-elem LDS rows.
// D# layout per CDNA5 ISA 8.3/8.4 (group0 128b, group1 256b).
// Toolchains disagree on builtin arity: ROCm 7.2 (clang-22) = 5 args,
// amdgpu-toolchain (clang-23+) = 6 args (extra int32x8 before cpol).
// ---------------------------------------------------------------------------
__device__ __forceinline__ void tdm_load_tile(const void* g, unsigned lds,
                                              unsigned tile_d0, unsigned tile_d1,
                                              unsigned long long td0_stride,
                                              unsigned tensor_d0,
                                              unsigned tensor_d1) {
  const unsigned long long ga = (unsigned long long)g;
  uint32x4 g0;
  g0[0] = 1u;                                        // count=1, user mode
  g0[1] = lds;                                       // lds_addr
  g0[2] = (unsigned)ga;                              // global_addr[31:0]
  g0[3] = (unsigned)((ga >> 32) & 0x01ffffffu) | (2u << 30);  // addr[56:32]|type=2
  int32x8 g1;
  g1[0] = (1 << 16) |                                // data_size = 2 bytes
          (1 << 20) |                                // pad_enable
          (3 << 22) |                                // pad_interval: 16 DWORDs
          (3 << 25);                                 // pad_amount: 4 DWORDs
  g1[1] = (int)((tensor_d0 & 0xffffu) << 16);        // tensor_dim0[15:0]
  g1[2] = (int)(((tensor_d0 >> 16) & 0xffffu) |      // tensor_dim0[31:16]
                ((tensor_d1 & 0xffffu) << 16));      // tensor_dim1[15:0]
  g1[3] = (int)(((tensor_d1 >> 16) & 0xffffu) |      // tensor_dim1[31:16]
                ((tile_d0 & 0xffffu) << 16));        // tile_dim0
  g1[4] = (int)(tile_d1 & 0xffffu);                  // tile_dim1 (tile_dim2=0)
  g1[5] = (int)(unsigned)td0_stride;                 // tensor_dim0_stride[31:0]
  g1[6] = (int)((td0_stride >> 32) & 0xffffu);       // tensor_dim0_stride[47:32]
  g1[7] = 0;
  const int32x4 gz4 = {0, 0, 0, 0};
  const int32x8 gz8 = {0, 0, 0, 0, 0, 0, 0, 0};
#if __clang_major__ >= 23
  __builtin_amdgcn_tensor_load_to_lds(g0, g1, gz4, gz4, gz8, 0);
#else
  __builtin_amdgcn_tensor_load_to_lds(g0, g1, gz4, gz4, 0);
#endif
}
#endif

// ---------------------------------------------------------------------------
// elementwise f32 -> bf16 convert (A-side operands: encoder)
// ---------------------------------------------------------------------------
__global__ void k_f32_to_bf16(const float* __restrict__ src,
                              __bf16* __restrict__ dst, int n) {
  int i = blockIdx.x * blockDim.x + threadIdx.x;
  if (i < n) dst[i] = (__bf16)src[i];
}

// ---------------------------------------------------------------------------
// weight convert f32[K,N] -> bf16 panel layout [K/32][N][32]
// ---------------------------------------------------------------------------
__global__ void k_weight_to_panel(const float* __restrict__ W,
                                  __bf16* __restrict__ Wp, int K, int N) {
  int i = blockIdx.x * blockDim.x + threadIdx.x;
  if (i >= K * N) return;
  const int k = i / N;
  const int n = i - k * N;
  Wp[(size_t)(k >> 5) * N * 32 + (size_t)n * 32 + (k & 31)] = (__bf16)W[i];
}

// ---------------------------------------------------------------------------
// LayerNorm over rows of 1024; emits fp32 (residual path) and bf16 (GEMM path)
// ---------------------------------------------------------------------------
__global__ void __launch_bounds__(256)
k_layernorm(const float* __restrict__ x, const float* __restrict__ g,
            const float* __restrict__ b, float* __restrict__ out32,
            __bf16* __restrict__ out16) {
  __shared__ float s_sum[256];
  __shared__ float s_sq[256];
  const int row = blockIdx.x;
  const float* xr = x + (size_t)row * D_MODEL;
  const int base = threadIdx.x * 4;
  float v[4];
  float lsum = 0.0f, lsq = 0.0f;
#pragma unroll
  for (int i = 0; i < 4; ++i) {
    v[i] = xr[base + i];
    lsum += v[i];
    lsq += v[i] * v[i];
  }
  s_sum[threadIdx.x] = lsum;
  s_sq[threadIdx.x] = lsq;
  __syncthreads();
  for (int off = 128; off > 0; off >>= 1) {
    if (threadIdx.x < off) {
      s_sum[threadIdx.x] += s_sum[threadIdx.x + off];
      s_sq[threadIdx.x] += s_sq[threadIdx.x + off];
    }
    __syncthreads();
  }
  const float mean = s_sum[0] * (1.0f / D_MODEL);
  const float var = s_sq[0] * (1.0f / D_MODEL) - mean * mean;
  const float rstd = rsqrtf(var + 1e-5f);
#pragma unroll
  for (int i = 0; i < 4; ++i) {
    float y = (v[i] - mean) * rstd * g[base + i] + b[base + i];
    out32[(size_t)row * D_MODEL + base + i] = y;
    out16[(size_t)row * D_MODEL + base + i] = (__bf16)y;
  }
}

// ---------------------------------------------------------------------------
// Tiled bf16 GEMM: C[M,N] = A[M,K] @ Bpanel[K/32,N,32] + bias (+GELU)(+resid)
// 256 threads = 8 waves, 128x128 tile, BK=32, double-buffered LDS.
// TDM path: wave 0 issues TENSOR_LOAD_TO_LDS for tile k+1 (hardware-padded
// 40-elem rows) while all waves run WMMA on tile k; s_wait_tensorcnt + barrier.
// Fallback path: per-thread b128 staging with global prefetch.
// ---------------------------------------------------------------------------
__global__ void __launch_bounds__(256)
k_gemm_bf16(const __bf16* __restrict__ A, const __bf16* __restrict__ Bp,
            const float* __restrict__ bias, const float* __restrict__ resid,
            float* __restrict__ out32, __bf16* __restrict__ out16,
            int M, int N, int K, int gelu) {
  __shared__ __align__(16) __bf16 As[2][128][40];  // [row][k], 20-dword stride
  __shared__ __align__(16) __bf16 Bs[2][128][40];  // [col][k], 20-dword stride

  const int tid  = threadIdx.x;
  const int wid  = tid >> 5;
  const int lane = tid & 31;
  const int l16  = lane & 15;
  const int hi   = lane >> 4;      // 0 = lanes 0-15, 1 = lanes 16-31
  const int row0 = blockIdx.y * 128;
  const int col0 = blockIdx.x * 128;
  const int rgrp = wid >> 2;       // 0..1 -> 64-row group
  const int cgrp = wid & 3;        // 0..3 -> 32-col group

  floatx8 acc[4][2];
#pragma unroll
  for (int mi = 0; mi < 4; ++mi)
#pragma unroll
    for (int ni = 0; ni < 2; ++ni) acc[mi][ni] = zero8();

  const size_t bstep = (size_t)N * 32;   // panel stride per K-step
  const int nkp = K >> 5;

#if HAVE_TDM
  const __bf16* Abase = A + (size_t)row0 * K;
  const __bf16* Bbase = Bp + (size_t)col0 * 32;
  // prologue: DMA tile 0 into buffer 0
  if (wid == 0) {
    tdm_load_tile(Abase, (unsigned)(unsigned long long)&As[0][0][0],
                  32, 128, (unsigned long long)K, (unsigned)K, (unsigned)M);
    tdm_load_tile(Bbase, (unsigned)(unsigned long long)&Bs[0][0][0],
                  4096, 1, 4096ull, (unsigned)(N * 32), 1u);
    __builtin_amdgcn_s_wait_tensorcnt(0);
  }
  __syncthreads();
#else
  // staging: each thread moves 16 bf16 of A and 16 bf16 of B per K-step
  const int srow = tid >> 1;             // 0..127 (A row / B col)
  const int sseg = (tid & 1) * 16;       // 0 or 16
  const __bf16* Apt = A + (size_t)(row0 + srow) * K + sseg;
  const __bf16* Bpt = Bp + (size_t)(col0 + srow) * 32 + sseg;
  uint4 ra0, ra1, rb0, rb1;
  ra0 = ((const uint4*)Apt)[0];
  ra1 = ((const uint4*)Apt)[1];
  rb0 = ((const uint4*)Bpt)[0];
  rb1 = ((const uint4*)Bpt)[1];
  *(uint4*)&As[0][srow][sseg]     = ra0;
  *(uint4*)&As[0][srow][sseg + 8] = ra1;
  *(uint4*)&Bs[0][srow][sseg]     = rb0;
  *(uint4*)&Bs[0][srow][sseg + 8] = rb1;
  __syncthreads();
#endif

  int cur = 0;
  for (int kp = 0; kp < nkp; ++kp) {
    const bool more = (kp + 1) < nkp;
#if HAVE_TDM
    if (more && wid == 0) {
      tdm_load_tile(Abase + (size_t)(kp + 1) * 32,
                    (unsigned)(unsigned long long)&As[cur ^ 1][0][0],
                    32, 128, (unsigned long long)K, (unsigned)K, (unsigned)M);
      tdm_load_tile(Bbase + (size_t)(kp + 1) * bstep,
                    (unsigned)(unsigned long long)&Bs[cur ^ 1][0][0],
                    4096, 1, 4096ull, (unsigned)(N * 32), 1u);
    }
#else
    if (more) {
      const __bf16* an = Apt + (size_t)(kp + 1) * 32;
      const __bf16* bn = Bpt + (size_t)(kp + 1) * bstep;
      ra0 = ((const uint4*)an)[0];
      ra1 = ((const uint4*)an)[1];
      rb0 = ((const uint4*)bn)[0];
      rb1 = ((const uint4*)bn)[1];
      if (kp + 2 < nkp) {
        __builtin_prefetch(Apt + (size_t)(kp + 2) * 32, 0, 0);
        __builtin_prefetch(Bpt + (size_t)(kp + 2) * bstep, 0, 0);
      }
    }
#endif

    // A fragments: lane<16 holds K 0..15; lane>=16 holds 16..31
    bf16x16 afr[4];
    const int kb2 = hi * 16;
#pragma unroll
    for (int mi = 0; mi < 4; ++mi) {
      const int r = rgrp * 64 + mi * 16 + l16;
#pragma unroll
      for (int i = 0; i < 8; ++i) {
        afr[mi][i]     = As[cur][r][kb2 + i];
        afr[mi][i + 8] = As[cur][r][8 + kb2 + i];
      }
    }
    // B fragments: contiguous 16 bf16 per lane from the [col][k] panel
    bf16x16 bfr[2];
    const int rb = hi * 16;
#pragma unroll
    for (int ni = 0; ni < 2; ++ni) {
      const int c = cgrp * 32 + ni * 16 + l16;
#pragma unroll
      for (int i = 0; i < 16; ++i) bfr[ni][i] = Bs[cur][c][rb + i];
    }
#pragma unroll
    for (int mi = 0; mi < 4; ++mi)
#pragma unroll
      for (int ni = 0; ni < 2; ++ni)
        acc[mi][ni] = __builtin_amdgcn_wmma_f32_16x16x32_bf16(
            false, afr[mi], false, bfr[ni], (short)0, acc[mi][ni], false, false);

#if HAVE_TDM
    if (more && wid == 0) __builtin_amdgcn_s_wait_tensorcnt(0);
#else
    if (more) {
      *(uint4*)&As[cur ^ 1][srow][sseg]     = ra0;
      *(uint4*)&As[cur ^ 1][srow][sseg + 8] = ra1;
      *(uint4*)&Bs[cur ^ 1][srow][sseg]     = rb0;
      *(uint4*)&Bs[cur ^ 1][srow][sseg + 8] = rb1;
    }
#endif
    __syncthreads();
    cur ^= 1;
  }

  // epilogue: C/D layout -> VGPR r: rows r (lanes 0-15) and r+8 (lanes 16-31)
#pragma unroll
  for (int mi = 0; mi < 4; ++mi) {
#pragma unroll
    for (int ni = 0; ni < 2; ++ni) {
      const int c = col0 + cgrp * 32 + ni * 16 + l16;
      const float bv = bias[c];
#pragma unroll
      for (int r = 0; r < 8; ++r) {
        const int rr = row0 + rgrp * 64 + mi * 16 + r + hi * 8;
        float v = acc[mi][ni][r] + bv;
        if (gelu) v = 0.5f * v * (1.0f + erff(v * 0.70710678118654752f));
        const size_t idx = (size_t)rr * N + c;
        if (resid) v += resid[idx];
        if (out32) out32[idx] = v;
        if (out16) out16[idx] = (__bf16)v;
      }
    }
  }
}

// ---------------------------------------------------------------------------
// Flash attention (bf16 QK^T + PV via WMMA, fp32 online softmax).
// Q,K,V,O are [B, S, D] with head h at columns h*64..h*64+63.
// Block = 128 threads = 4 waves; each wave owns 16 query rows.
// V is staged transposed (Vt[hd][key]) so PV B-fragments are contiguous.
// ---------------------------------------------------------------------------
template <bool CAUSAL>
__global__ void __launch_bounds__(128)
k_attention(const __bf16* __restrict__ Q, const __bf16* __restrict__ Kt,
            const __bf16* __restrict__ V, __bf16* __restrict__ O) {
  __shared__ __align__(16) __bf16 Kl[32][72];  // [key][hd]
  __shared__ __align__(16) __bf16 Vt[64][40];  // [hd][key]
  __shared__ __align__(16) __bf16 Pl[64][40];  // per-wave P transpose scratch

  const int tid  = threadIdx.x;
  const int wid  = tid >> 5;
  const int lane = tid & 31;
  const int l16  = lane & 15;
  const int hi   = lane >> 4;
  const int bh   = blockIdx.y;
  const int b    = bh >> 4;
  const int h    = bh & 15;
  const int qblk = blockIdx.x * 64;
  const int qbase = qblk + wid * 16;
  const size_t colbase = (size_t)h * HDIM;

  // Q fragments for both 32-wide K-steps of HDIM=64 (kept in registers)
  bf16x16 aq[2];
  {
    const __bf16* qrow = Q + (size_t)(b * SEQ + qbase + l16) * D_MODEL + colbase;
    const int kb2 = hi * 16;
#pragma unroll
    for (int ks = 0; ks < 2; ++ks) {
#pragma unroll
      for (int i = 0; i < 8; ++i) {
        aq[ks][i]     = qrow[ks * 32 + kb2 + i];
        aq[ks][i + 8] = qrow[ks * 32 + 8 + kb2 + i];
      }
    }
  }

  floatx8 oacc[4];
#pragma unroll
  for (int t = 0; t < 4; ++t) oacc[t] = zero8();
  float m[8], l[8];
#pragma unroll
  for (int r = 0; r < 8; ++r) { m[r] = -__builtin_inff(); l[r] = 0.0f; }

  const int nkb = CAUSAL ? (qblk / 32 + 2) : (SEQ / 32);
  const float scale = 0.125f;  // 1/sqrt(64)

  for (int kb = 0; kb < nkb; ++kb) {
    // stage 32x64 K (row-major) and V (transposed) blocks
    {
      const int idx = tid * 16;       // 2048 bf16, 16 per thread
      const int vr  = idx >> 6;       // key row 0..31
      const int vc  = idx & 63;       // hd col 0,16,32,48
      const size_t grow =
          (size_t)(b * SEQ + kb * 32 + vr) * D_MODEL + colbase + vc;
      const uint4* gk = (const uint4*)(Kt + grow);
      *(uint4*)&Kl[vr][vc]     = gk[0];
      *(uint4*)&Kl[vr][vc + 8] = gk[1];
      const uint4* gv = (const uint4*)(V + grow);
      uint4 va = gv[0], vb = gv[1];
      unsigned w[8] = {va.x, va.y, va.z, va.w, vb.x, vb.y, vb.z, vb.w};
#pragma unroll
      for (int j = 0; j < 8; ++j) {
        Vt[vc + 2 * j][vr]     = bf_from_bits((unsigned short)(w[j] & 0xffff));
        Vt[vc + 2 * j + 1][vr] = bf_from_bits((unsigned short)(w[j] >> 16));
      }
    }
    __syncthreads();

    // scores S(16x32) = Q(16x64) @ K^T(64x32); two 16-col tiles, two K-steps
    floatx8 s0 = zero8(), s1 = zero8();
    const int rb = hi * 16;
#pragma unroll
    for (int ks = 0; ks < 2; ++ks) {
      bf16x16 b0, b1;
#pragma unroll
      for (int i = 0; i < 16; ++i) {
        b0[i] = Kl[l16][ks * 32 + rb + i];        // B[hd][key] = K[key][hd]
        b1[i] = Kl[16 + l16][ks * 32 + rb + i];
      }
      s0 = __builtin_amdgcn_wmma_f32_16x16x32_bf16(false, aq[ks], false, b0,
                                                   (short)0, s0, false, false);
      s1 = __builtin_amdgcn_wmma_f32_16x16x32_bf16(false, aq[ks], false, b1,
                                                   (short)0, s1, false, false);
    }

    // online softmax update (rows live in lane halves; reduce over 16 lanes)
#pragma unroll
    for (int r = 0; r < 8; ++r) {
      const int qrow = qbase + r + hi * 8;
      float v0 = s0[r] * scale;
      float v1 = s1[r] * scale;
      if (CAUSAL) {
        const int c0 = kb * 32 + l16;
        if (c0 > qrow) v0 = -__builtin_inff();
        if (c0 + 16 > qrow) v1 = -__builtin_inff();
      }
      float mx = fmaxf(v0, v1);
#pragma unroll
      for (int d = 1; d < 16; d <<= 1) mx = fmaxf(mx, __shfl_xor(mx, d, 32));
      const float mnew = fmaxf(m[r], mx);
      const float alpha = __expf(m[r] - mnew);
      const float p0 = __expf(v0 - mnew);
      const float p1 = __expf(v1 - mnew);
      float rs = p0 + p1;
#pragma unroll
      for (int d = 1; d < 16; d <<= 1) rs += __shfl_xor(rs, d, 32);
      l[r] = l[r] * alpha + rs;
      m[r] = mnew;
#pragma unroll
      for (int t = 0; t < 4; ++t) oacc[t][r] *= alpha;
      // store P (C-layout) to wave-private LDS for transpose to A-layout
      Pl[wid * 16 + r + hi * 8][l16]      = (__bf16)p0;
      Pl[wid * 16 + r + hi * 8][16 + l16] = (__bf16)p1;
    }

    // P(16x32) @ V(32x64): per-wave LDS ops are in-order, no barrier needed
    bf16x16 ap;
    {
      const int kb2 = hi * 16;
#pragma unroll
      for (int i = 0; i < 8; ++i) {
        ap[i]     = Pl[wid * 16 + l16][kb2 + i];
        ap[i + 8] = Pl[wid * 16 + l16][8 + kb2 + i];
      }
    }
#pragma unroll
    for (int nt = 0; nt < 4; ++nt) {
      bf16x16 bv;
#pragma unroll
      for (int i = 0; i < 16; ++i) bv[i] = Vt[nt * 16 + l16][rb + i];
      oacc[nt] = __builtin_amdgcn_wmma_f32_16x16x32_bf16(false, ap, false, bv,
                                                         (short)0, oacc[nt],
                                                         false, false);
    }
    __syncthreads();
  }

  // normalize and store bf16 output (consumed by the output projection GEMM)
#pragma unroll
  for (int nt = 0; nt < 4; ++nt) {
#pragma unroll
    for (int r = 0; r < 8; ++r) {
      const int qrow = qbase + r + hi * 8;
      const float ov = oacc[nt][r] / l[r];
      O[(size_t)(b * SEQ + qrow) * D_MODEL + colbase + nt * 16 + l16] = (__bf16)ov;
    }
  }
}

// ---------------------------------------------------------------------------
// host launch
// ---------------------------------------------------------------------------
extern "C" void kernel_launch(void* const* d_in, const int* in_sizes, int n_in,
                              void* d_out, int out_size, void* d_ws,
                              size_t ws_size, hipStream_t stream) {
  const float* decoder = (const float*)d_in[0];
  const float* encoder = (const float*)d_in[1];
  const float* sa_wq = (const float*)d_in[2];
  const float* sa_bq = (const float*)d_in[3];
  const float* sa_wk = (const float*)d_in[4];
  const float* sa_bk = (const float*)d_in[5];
  const float* sa_wv = (const float*)d_in[6];
  const float* sa_bv = (const float*)d_in[7];
  const float* sa_wo = (const float*)d_in[8];
  const float* sa_bo = (const float*)d_in[9];
  const float* ca_wq = (const float*)d_in[10];
  const float* ca_bq = (const float*)d_in[11];
  const float* ca_wk = (const float*)d_in[12];
  const float* ca_bk = (const float*)d_in[13];
  const float* ca_wv = (const float*)d_in[14];
  const float* ca_bv = (const float*)d_in[15];
  const float* ca_wo = (const float*)d_in[16];
  const float* ca_bo = (const float*)d_in[17];
  const float* ln1_g = (const float*)d_in[18];
  const float* ln1_b = (const float*)d_in[19];
  const float* ln2_g = (const float*)d_in[20];
  const float* ln2_b = (const float*)d_in[21];
  const float* ln3_g = (const float*)d_in[22];
  const float* ln3_b = (const float*)d_in[23];
  const float* ffn_w1 = (const float*)d_in[24];
  const float* ffn_b1 = (const float*)d_in[25];
  const float* ffn_w2 = (const float*)d_in[26];
  const float* ffn_b2 = (const float*)d_in[27];
  float* out = (float*)d_out;

  const size_t ME = (size_t)MROWS * D_MODEL;          // 8M activation elems
  const size_t WE = (size_t)D_MODEL * D_MODEL;        // 1M per attn weight

  // workspace carve-up
  char* ws = (char*)d_ws;
  size_t o = 0;
  auto take = [&](size_t bytes) {
    void* p = ws + o;
    o += (bytes + 255) & ~(size_t)255;
    return p;
  };
  __bf16* w_bf   = (__bf16*)take(16u * 1024 * 1024 * sizeof(__bf16)); // 10 mats
  __bf16* enc_bf = (__bf16*)take(ME * sizeof(__bf16));
  __bf16* act_bf = (__bf16*)take(ME * sizeof(__bf16));   // x/y/z bf16 (reused)
  __bf16* q_bf   = (__bf16*)take(ME * sizeof(__bf16));
  __bf16* k_bf   = (__bf16*)take(ME * sizeof(__bf16));
  __bf16* v_bf   = (__bf16*)take(ME * sizeof(__bf16));
  __bf16* rep_bf = (__bf16*)take(ME * sizeof(__bf16));
  __bf16* h_bf   = (__bf16*)take((size_t)MROWS * FFDIM * sizeof(__bf16));
  float*  cur32  = (float*)take(ME * sizeof(float));     // x/y/z fp32 (reused)

  __bf16* w_sa_q = w_bf + 0 * WE;
  __bf16* w_sa_k = w_bf + 1 * WE;
  __bf16* w_sa_v = w_bf + 2 * WE;
  __bf16* w_sa_o = w_bf + 3 * WE;
  __bf16* w_ca_q = w_bf + 4 * WE;
  __bf16* w_ca_k = w_bf + 5 * WE;
  __bf16* w_ca_v = w_bf + 6 * WE;
  __bf16* w_ca_o = w_bf + 7 * WE;
  __bf16* w_ff1  = w_bf + 8 * WE;                       // D x FF panels (4M)
  __bf16* w_ff2  = w_ff1 + (size_t)D_MODEL * FFDIM;     // FF x D panels (4M)

  auto convw = [&](const float* s, __bf16* d, int K, int N) {
    const size_t n = (size_t)K * N;
    k_weight_to_panel<<<dim3((unsigned)((n + 255) / 256)), dim3(256), 0,
                        stream>>>(s, d, K, N);
  };
  convw(sa_wq, w_sa_q, D_MODEL, D_MODEL);
  convw(sa_wk, w_sa_k, D_MODEL, D_MODEL);
  convw(sa_wv, w_sa_v, D_MODEL, D_MODEL);
  convw(sa_wo, w_sa_o, D_MODEL, D_MODEL);
  convw(ca_wq, w_ca_q, D_MODEL, D_MODEL);
  convw(ca_wk, w_ca_k, D_MODEL, D_MODEL);
  convw(ca_wv, w_ca_v, D_MODEL, D_MODEL);
  convw(ca_wo, w_ca_o, D_MODEL, D_MODEL);
  convw(ffn_w1, w_ff1, D_MODEL, FFDIM);
  convw(ffn_w2, w_ff2, FFDIM, D_MODEL);
  k_f32_to_bf16<<<dim3((unsigned)((ME + 255) / 256)), dim3(256), 0, stream>>>(
      encoder, enc_bf, (int)ME);

  const dim3 blk256(256), blk128(128);
  const dim3 gLN(MROWS);
  const dim3 gGemmDD(D_MODEL / 128, MROWS / 128);   // (8, 64)
  const dim3 gGemmDF(FFDIM / 128, MROWS / 128);     // (32, 64)
  const dim3 gAttn(SEQ / 64, BATCH * NHEAD);        // (16, 128)

  // x = LN1(decoder)
  k_layernorm<<<gLN, blk256, 0, stream>>>(decoder, ln1_g, ln1_b, cur32, act_bf);
  // self-attention Q/K/V projections (bf16 out)
  k_gemm_bf16<<<gGemmDD, blk256, 0, stream>>>(act_bf, w_sa_q, sa_bq, nullptr,
                                              nullptr, q_bf, MROWS, D_MODEL,
                                              D_MODEL, 0);
  k_gemm_bf16<<<gGemmDD, blk256, 0, stream>>>(act_bf, w_sa_k, sa_bk, nullptr,
                                              nullptr, k_bf, MROWS, D_MODEL,
                                              D_MODEL, 0);
  k_gemm_bf16<<<gGemmDD, blk256, 0, stream>>>(act_bf, w_sa_v, sa_bv, nullptr,
                                              nullptr, v_bf, MROWS, D_MODEL,
                                              D_MODEL, 0);
  // causal flash attention
  k_attention<true><<<gAttn, blk128, 0, stream>>>(q_bf, k_bf, v_bf, rep_bf);
  // x = x + rep @ Wo + bo   (in-place residual into cur32)
  k_gemm_bf16<<<gGemmDD, blk256, 0, stream>>>(rep_bf, w_sa_o, sa_bo, cur32,
                                              cur32, nullptr, MROWS, D_MODEL,
                                              D_MODEL, 0);
  // y = LN2(x)
  k_layernorm<<<gLN, blk256, 0, stream>>>(cur32, ln2_g, ln2_b, cur32, act_bf);
  // cross-attention projections (Q from y, K/V from raw encoder)
  k_gemm_bf16<<<gGemmDD, blk256, 0, stream>>>(act_bf, w_ca_q, ca_bq, nullptr,
                                              nullptr, q_bf, MROWS, D_MODEL,
                                              D_MODEL, 0);
  k_gemm_bf16<<<gGemmDD, blk256, 0, stream>>>(enc_bf, w_ca_k, ca_bk, nullptr,
                                              nullptr, k_bf, MROWS, D_MODEL,
                                              D_MODEL, 0);
  k_gemm_bf16<<<gGemmDD, blk256, 0, stream>>>(enc_bf, w_ca_v, ca_bv, nullptr,
                                              nullptr, v_bf, MROWS, D_MODEL,
                                              D_MODEL, 0);
  // non-causal flash attention
  k_attention<false><<<gAttn, blk128, 0, stream>>>(q_bf, k_bf, v_bf, rep_bf);
  // y = y + rep @ Wo + bo
  k_gemm_bf16<<<gGemmDD, blk256, 0, stream>>>(rep_bf, w_ca_o, ca_bo, cur32,
                                              cur32, nullptr, MROWS, D_MODEL,
                                              D_MODEL, 0);
  // z = LN3(y)
  k_layernorm<<<gLN, blk256, 0, stream>>>(cur32, ln3_g, ln3_b, cur32, act_bf);
  // h = gelu(z @ W1 + b1)  (bf16 out)
  k_gemm_bf16<<<gGemmDF, blk256, 0, stream>>>(act_bf, w_ff1, ffn_b1, nullptr,
                                              nullptr, h_bf, MROWS, FFDIM,
                                              D_MODEL, 1);
  // out = z + h @ W2 + b2  (fp32 out)
  k_gemm_bf16<<<gGemmDD, blk256, 0, stream>>>(h_bf, w_ff2, ffn_b2, cur32, out,
                                              nullptr, MROWS, D_MODEL, FFDIM,
                                              0);
}
